// HarmonicRenderer_65085934403676
// MI455X (gfx1250) — compile-verified, matrix operand order checked by reading, not run on hardware
//
#include <hip/hip_runtime.h>

typedef __attribute__((ext_vector_type(16))) _Float16 v16h;
typedef __attribute__((ext_vector_type(8)))  float    v8f;

#define NUM_NOTES 82
#define KPAD      96
#define KHARM     400
#define NBINS     1024
#define NMELS     256
#define MROWS     (32*4096)

// ---------------------------------------------------------------------------
// Kernel 1: cols[n][f] = sum_k omega*mask * exp(-0.5*((f - f0*k)/std)^2)
// grid (82, 4) x 256 threads; LDS-staged per-note omega*mask and 1/std.
// ---------------------------------------------------------------------------
__global__ void __launch_bounds__(256)
cols_kernel(const float* __restrict__ omega,
            const float* __restrict__ stdv,
            const float* __restrict__ mask,
            float* __restrict__ cols) {
    __shared__ float s_om[KHARM];
    __shared__ float s_rs[KHARM];
    const int n = blockIdx.x;
    const int t = threadIdx.x;
    const float* om = omega + n * KHARM;
    const float* sd = stdv  + n * KHARM;
    const float* mk = mask  + n * KHARM;
    for (int i = t; i < KHARM; i += 256) {
        s_om[i] = om[i] * mk[i];
        s_rs[i] = 1.0f / sd[i];
    }
    __syncthreads();

    const int j  = blockIdx.y * 256 + t;                 // freq bin 0..1023
    const float f  = (float)j * (11025.0f / 1023.0f);    // linspace(0, 11025, 1024)
    const float f0 = 27.5f * exp2f((float)n * (1.0f / 12.0f));

    float acc = 0.0f;
    for (int k = 0; k < KHARM; ++k) {
        float c   = f0 * (float)(k + 1);
        float tt  = (f - c) * s_rs[k];
        float arg = -0.5f * tt * tt;
        if (arg > -45.0f)                                // exp underflows anyway
            acc += s_om[k] * __expf(arg);
    }
    cols[n * NBINS + j] = acc;
}

// ---------------------------------------------------------------------------
// Kernel 2: spec = cols @ MEL^T (slaney mel computed analytically), then
// w = (clip(log(spec+eps)) - XMIN)/(XMAX-XMIN), packed as f16 into the
// WMMA B-fragment layout (K padded 82->96 with zeros).
// grid 96 (k) x 256 (m) threads.
// ---------------------------------------------------------------------------
__global__ void __launch_bounds__(256)
w_pack_kernel(const float* __restrict__ cols, _Float16* __restrict__ bpack) {
    __shared__ float s_cols[NBINS];
    const int k = blockIdx.x;    // note (K index), 0..95
    const int m = threadIdx.x;   // mel (N index), 0..255

    if (k < NUM_NOTES) {
        const float* cr = cols + k * NBINS;
        for (int i = m; i < NBINS; i += 256) s_cols[i] = cr[i];
    }
    __syncthreads();

    float w = 0.0f;
    if (k < NUM_NOTES) {
        // slaney mel points: mel_f[i] = mel_to_hz(i * melmax/257)
        const float melmax = 15.0f + 27.0f * logf(11.025f) / logf(6.4f);
        const float dm     = melmax / 257.0f;
        const float lstep  = logf(6.4f) / 27.0f;
        float mf[3];
        #pragma unroll
        for (int i = 0; i < 3; ++i) {
            float mm = dm * (float)(m + i);
            mf[i] = (mm < 15.0f) ? mm * (200.0f / 3.0f)
                                 : 1000.0f * __expf(lstep * (mm - 15.0f));
        }
        const float inv_d0 = 1.0f / (mf[1] - mf[0]);
        const float inv_d1 = 1.0f / (mf[2] - mf[1]);
        const float enorm  = 2.0f / (mf[2] - mf[0]);

        float acc = 0.0f;
        for (int j = 0; j < NBINS; ++j) {
            // MEL column j uses fftfreqs[j+1] = (j+1)*11025/1024 (the [:,1:] slice)
            float fq = (float)(j + 1) * (11025.0f / 1024.0f);
            float lo = (fq - mf[0]) * inv_d0;
            float up = (mf[2] - fq) * inv_d1;
            float wg = fminf(lo, up);
            wg = fmaxf(wg, 0.0f);
            acc += s_cols[j] * wg;
        }
        const float XMIN = -13.8155105579f;   // log(1e-6)
        const float XMAX = 3.0f;
        float x = logf(acc * enorm + 1e-6f);
        x = fminf(fmaxf(x, XMIN), XMAX);
        w = (x - XMIN) * (1.0f / (XMAX - XMIN));
    }

    // scatter into WMMA B-fragment layout: lane holds N=lane%16,
    // K(e) = 32*kb + e + 8*((e>>3) + (lane>=16))
    const int nt   = m >> 4, ll = m & 15;
    const int kb   = k >> 5, kr = k & 31;
    const int hi   = (kr >> 3) & 1;
    const int e    = ((kr >> 4) << 3) | (kr & 7);
    const int lane = hi * 16 + ll;
    bpack[(size_t)(((nt * 3 + kb) * 32) + lane) * 16 + e] = (_Float16)w;
}

// ---------------------------------------------------------------------------
// Kernel 3: out = y @ w via v_wmma_f32_16x16x32_f16.
// 128 threads = 4 waves; each wave owns a 16-row M-tile, all 16 N-tiles.
// B (48 KB packed fragments) coop-loaded into LDS once per workgroup.
// ---------------------------------------------------------------------------
__global__ void __launch_bounds__(128)
gemm_kernel(const float* __restrict__ y,
            const _Float16* __restrict__ bpack,
            float* __restrict__ out) {
    __shared__ v16h Blds[16 * 3 * 32];   // 48 KB

    const int tid = threadIdx.x;
    {   // cooperative load: 3072 uint4 = 48 KB
        const uint4* src = (const uint4*)bpack;
        uint4*       dst = (uint4*)Blds;
        #pragma unroll
        for (int i = 0; i < 24; ++i) dst[tid + i * 128] = src[tid + i * 128];
    }
    __syncthreads();

    const int wave = tid >> 5;
    const int lane = tid & 31;
    const int hi   = lane >> 4;
    const int mr   = lane & 15;
    const int m0   = (blockIdx.x * 4 + wave) * 16;
    const float* yr = y + (size_t)(m0 + mr) * NUM_NOTES;

    // A fragments: a[kb][e] = y[row][K], K = 32*kb + e + 8*((e>>3)+hi)
    v16h a[3];
    #pragma unroll
    for (int kb = 0; kb < 2; ++kb) {
        const int K0 = 32 * kb + 8 * hi;       // e = 0..7  -> K0..K0+7
        const int K1 = K0 + 16;                // e = 8..15 -> K1..K1+7
        #pragma unroll
        for (int p = 0; p < 4; ++p) {
            float2 v0 = *(const float2*)(yr + K0 + 2 * p);
            a[kb][2 * p + 0] = (_Float16)v0.x;
            a[kb][2 * p + 1] = (_Float16)v0.y;
            float2 v1 = *(const float2*)(yr + K1 + 2 * p);
            a[kb][8 + 2 * p + 0] = (_Float16)v1.x;
            a[kb][8 + 2 * p + 1] = (_Float16)v1.y;
        }
    }
    {   // kb = 2: K 64..79 fully valid; K 80..95 guarded (K < 82), pad = 0
        const int K0 = 64 + 8 * hi;
        #pragma unroll
        for (int p = 0; p < 4; ++p) {
            float2 v0 = *(const float2*)(yr + K0 + 2 * p);
            a[2][2 * p + 0] = (_Float16)v0.x;
            a[2][2 * p + 1] = (_Float16)v0.y;
        }
        const int K1 = 80 + 8 * hi;
        #pragma unroll
        for (int e = 0; e < 8; ++e) {
            int K = K1 + e;
            a[2][8 + e] = (K < NUM_NOTES) ? (_Float16)yr[K] : (_Float16)0.0f;
        }
    }

    #pragma unroll
    for (int nt = 0; nt < 16; ++nt) {
        v8f c = {0.f, 0.f, 0.f, 0.f, 0.f, 0.f, 0.f, 0.f};
        #pragma unroll
        for (int kb = 0; kb < 3; ++kb) {
            v16h b = Blds[(nt * 3 + kb) * 32 + lane];
            c = __builtin_amdgcn_wmma_f32_16x16x32_f16(
                    false, a[kb], false, b, (short)0, c, false, false);
        }
        const int col = nt * 16 + mr;
        #pragma unroll
        for (int r = 0; r < 8; ++r)          // D: VGPR r holds M = r + 8*hi
            out[(size_t)(m0 + r + 8 * hi) * NMELS + col] = c[r];
    }
}

// ---------------------------------------------------------------------------
extern "C" void kernel_launch(void* const* d_in, const int* in_sizes, int n_in,
                              void* d_out, int out_size, void* d_ws, size_t ws_size,
                              hipStream_t stream) {
    const float* y     = (const float*)d_in[0];   // (32,4096,82)
    const float* omega = (const float*)d_in[1];   // (82,400)
    const float* stdv  = (const float*)d_in[2];   // (82,400)
    const float* mask  = (const float*)d_in[3];   // (82,400)
    float* out = (float*)d_out;                   // (32,4096,256)

    float*     cols  = (float*)d_ws;                                        // 336 KB
    _Float16*  bpack = (_Float16*)((char*)d_ws + NUM_NOTES * NBINS * 4);    // 48 KB

    cols_kernel  <<<dim3(82, 4), 256, 0, stream>>>(omega, stdv, mask, cols);
    w_pack_kernel<<<KPAD,        256, 0, stream>>>(cols, bpack);
    gemm_kernel  <<<MROWS / 64,  128, 0, stream>>>(y, bpack, out);
}
